// DGR_57612691308898
// MI455X (gfx1250) — matrix operand order repaired and checked by reading, not compiled, not run on hardware
//
#include <hip/hip_runtime.h>

#define BB 8
#define CC 256
#define NN 9216          // 96*96
#define KK 64
#define OSTK (KK + 3*CC) // 832 stacked projection output channels
#define NSPLIT 8
#define NCHUNK (NN / NSPLIT)
#define BN_CNT ((float)((long)BB * NN))

typedef __attribute__((ext_vector_type(16))) __bf16 v16bf;
typedef __attribute__((ext_vector_type(8)))  float  v8f;
typedef __attribute__((ext_vector_type(4)))  unsigned u32x4;
typedef __attribute__((ext_vector_type(8)))  unsigned u32x8;

// ---------------- bf16 <-> f32 (RNE) ----------------
__device__ __forceinline__ __bf16 f2bf(float f) {
  union { float f; unsigned u; } a; a.f = f;
  unsigned r = a.u + 0x7FFFu + ((a.u >> 16) & 1u);
  unsigned short s = (unsigned short)(r >> 16);
  __bf16 h; __builtin_memcpy(&h, &s, 2); return h;
}
__device__ __forceinline__ float bf2f(__bf16 h) {
  unsigned short s; __builtin_memcpy(&s, &h, 2);
  union { unsigned u; float f; } a; a.u = ((unsigned)s) << 16; return a.f;
}

// ---------------- WMMA core ----------------
__device__ __forceinline__ v8f wmma_bf(v16bf a, v16bf b, v8f c) {
  return __builtin_amdgcn_wmma_f32_16x16x32_bf16(false, a, false, b, (short)0, c,
                                                 false, false);
}

// A tile 16x32 (MxK), storage row-major [M][lda], reduction contiguous.
__device__ __forceinline__ v16bf load_A16(const __bf16* A, long lda) {
  int lane = threadIdx.x & 31;
  const __bf16* p = A + (long)(lane & 15) * lda + ((lane >> 4) ? 8 : 0);
  v16bf r;
#pragma unroll
  for (int e = 0; e < 8; ++e) r[e] = p[e];
#pragma unroll
  for (int e = 0; e < 8; ++e) r[e + 8] = p[e + 16];
  return r;
}
__device__ __forceinline__ v16bf load_A16_f32(const float* A, long lda) {
  int lane = threadIdx.x & 31;
  const float* p = A + (long)(lane & 15) * lda + ((lane >> 4) ? 8 : 0);
  v16bf r;
#pragma unroll
  for (int e = 0; e < 8; ++e) r[e] = f2bf(p[e]);
#pragma unroll
  for (int e = 0; e < 8; ++e) r[e + 8] = f2bf(p[e + 16]);
  return r;
}
// B tile 32x16 (KxN), storage [ncol][ldb] with reduction contiguous.
__device__ __forceinline__ v16bf load_B16(const __bf16* Bp, long ldb) {
  int lane = threadIdx.x & 31;
  const __bf16* p = Bp + (long)(lane & 15) * ldb + ((lane >> 4) ? 8 : 0);
  v16bf r;
#pragma unroll
  for (int e = 0; e < 16; ++e) r[e] = p[e + ((e >> 3) << 3)];
  return r;
}
__device__ __forceinline__ v16bf load_B16_f32(const float* Bp, long ldb) {
  int lane = threadIdx.x & 31;
  const float* p = Bp + (long)(lane & 15) * ldb + ((lane >> 4) ? 8 : 0);
  v16bf r;
#pragma unroll
  for (int e = 0; e < 16; ++e) r[e] = f2bf(p[e + ((e >> 3) << 3)]);
  return r;
}

// ========== 1. x -> bf16 transposed copy  xbT[b][n][c] ==========
__global__ void dgr_xT(const float* __restrict__ x, __bf16* __restrict__ xbT) {
  long tot = (long)BB * CC * NN;
  for (long i = blockIdx.x * (long)blockDim.x + threadIdx.x; i < tot;
       i += (long)gridDim.x * blockDim.x) {
    int n = (int)(i % NN); long t = i / NN; int c = (int)(t % CC); int b = (int)(t / CC);
    xbT[((long)b * NN + n) * CC + c] = f2bf(x[i]);
  }
}

// ========== 2. pack stacked weights (832xC) + gcn_w^T ==========
__global__ void dgr_wpack(const float* wphi, const float* wth, const float* wro,
                          const float* wva, const float* gcnw,
                          __bf16* Wstk, __bf16* gwT) {
  int stride = gridDim.x * blockDim.x;
  int i0 = blockIdx.x * blockDim.x + threadIdx.x;
  for (int j = i0; j < OSTK * CC; j += stride) {
    int c = j % CC, o = j / CC; float v;
    if (o < KK)            v = wphi[o * CC + c];
    else if (o < KK + CC)  v = wth[(o - KK) * CC + c];
    else if (o < KK + 2*CC)v = wro[(o - KK - CC) * CC + c];
    else                   v = wva[(o - KK - 2*CC) * CC + c];
    Wstk[j] = f2bf(v);
  }
  for (int j = i0; j < CC * CC; j += stride) {
    int c = j % CC, o = j / CC;            // gwT[o][c] = gcnw[c][o]
    gwT[j] = f2bf(gcnw[c * CC + o]);
  }
}

// ========== 3. xmean[c] over (b,n) ==========
__global__ void dgr_xmean(const float* __restrict__ x, float* __restrict__ xmean) {
  int c = blockIdx.x, t = threadIdx.x;
  float s = 0.f;
  for (int i = t; i < BB * NN; i += 256) {
    int b = i / NN, n = i % NN;
    s += x[((long)b * CC + c) * NN + n];
  }
  __shared__ float red[256];
  red[t] = s; __syncthreads();
  for (int w = 128; w; w >>= 1) { if (t < w) red[t] += red[t + w]; __syncthreads(); }
  if (t == 0) xmean[c] = red[0] / BN_CNT;
}

// ========== 4. per-b covariance GEMM ==========
__global__ void dgr_cxx(const float* __restrict__ x, float* __restrict__ Cxxb) {
  int c1 = blockIdx.x * 16, c2 = blockIdx.y * 16, b = blockIdx.z;
  const float* xb = x + (long)b * CC * NN;
  v8f acc = {0.f,0.f,0.f,0.f,0.f,0.f,0.f,0.f};
  for (int n0 = 0; n0 < NN; n0 += 32) {
    v16bf a  = load_A16_f32(xb + (long)c1 * NN + n0, NN);
    v16bf bb = load_B16_f32(xb + (long)c2 * NN + n0, NN);
    acc = wmma_bf(a, bb, acc);
  }
  int lane = threadIdx.x & 31;
  int col = lane & 15, mo = (lane >> 4) ? 8 : 0;
#pragma unroll
  for (int r = 0; r < 8; ++r)
    Cxxb[((long)b * CC + c1 + r + mo) * CC + c2 + col] = acc[r];
}
__global__ void dgr_cxxred(const float* __restrict__ Cxxb, float* __restrict__ Cxx) {
  int i = blockIdx.x * blockDim.x + threadIdx.x;
  if (i >= CC * CC) return;
  float s = 0.f;
  for (int b = 0; b < BB; ++b) s += Cxxb[(long)b * CC * CC + i];
  Cxx[i] = s;
}

// ========== 5. analytic BN params for all 832 stacked channels ==========
__global__ void dgr_bnparams(const float* wphi, const float* wth, const float* wro,
                             const float* wva, const float* sphi, const float* bphi,
                             const float* sth, const float* bth, const float* sro,
                             const float* bro, const float* sva, const float* bva,
                             const float* __restrict__ Cxx, const float* __restrict__ xmean,
                             float* alpha, float* beta) {
  int o = blockIdx.x, t = threadIdx.x;
  const float* wrow; float sc, bi;
  if (o < KK)              { wrow = wphi + o * CC;                sc = sphi[o]; bi = bphi[o]; }
  else if (o < KK + CC)    { int u = o - KK;        wrow = wth + u * CC; sc = sth[u]; bi = bth[u]; }
  else if (o < KK + 2*CC)  { int u = o - KK - CC;   wrow = wro + u * CC; sc = sro[u]; bi = bro[u]; }
  else                     { int u = o - KK - 2*CC; wrow = wva + u * CC; sc = sva[u]; bi = bva[u]; }
  float wc = wrow[t];
  float mu_p = wc * xmean[t];
  float inner = 0.f;
  for (int c2 = 0; c2 < CC; ++c2) inner += Cxx[(long)t * CC + c2] * wrow[c2];
  float q_p = wc * inner;
  __shared__ float r1[256], r2[256];
  r1[t] = mu_p; r2[t] = q_p; __syncthreads();
  for (int w = 128; w; w >>= 1) { if (t < w) { r1[t] += r1[t + w]; r2[t] += r2[t + w]; } __syncthreads(); }
  if (t == 0) {
    float mu = r1[0];
    float var = r2[0] / BN_CNT - mu * mu;
    if (var < 0.f) var = 0.f;
    float a = sc * rsqrtf(var + 1e-5f);
    alpha[o] = a; beta[o] = bi - mu * a;
  }
}

// ========== 6. fused projection GEMM + BN + ReLU, TDM-staged B panel ==========
// Block: 128 threads (4 waves), one 16-wide n tile, 4 consecutive 16-wide o tiles.
// The 16x256 bf16 x panel (8KB) is DMA'd to LDS once per block via the Tensor
// Data Mover, then all 4 waves run the full K loop out of LDS.
__global__ void dgr_proj(const __bf16* __restrict__ Wstk, const __bf16* __restrict__ xbT,
                         const float* __restrict__ alpha, const float* __restrict__ beta,
                         __bf16* __restrict__ act) {
  __shared__ __bf16 smem[16 * CC];   // [n][c] panel, 8 KB
  int b = blockIdx.z;
  int n0 = blockIdx.y * 16;
  int o0 = (blockIdx.x * 4 + (threadIdx.x >> 5)) * 16;

  if ((threadIdx.x >> 5) == 0) {
    unsigned long gaddr = (unsigned long)(const void*)(xbT + ((long)b * NN + n0) * CC);
    unsigned lds = (unsigned)(unsigned long)(void*)smem;
    u32x4 g0;
    g0[0] = 1u;                                                   // count=1, user mode
    g0[1] = lds;                                                  // lds_addr
    g0[2] = (unsigned)gaddr;                                      // global_addr[31:0]
    g0[3] = (unsigned)((gaddr >> 32) & 0x1FFFFFFu) | (2u << 30);  // addr[56:32] | type=2
    u32x8 g1;
    g1[0] = 1u << 16;                    // data_size = 2 bytes
    g1[1] = ((unsigned)CC) << 16;        // tensor_dim0[15:0] = 256
    g1[2] = 16u << 16;                   // tensor_dim0[31:16]=0 | tensor_dim1[15:0]=16
    g1[3] = ((unsigned)CC) << 16;        // tensor_dim1[31:16]=0 | tile_dim0 = 256
    g1[4] = 16u;                         // tile_dim1 = 16, tile_dim2 = 0
    g1[5] = (unsigned)CC;                // tensor_dim0_stride = 256 (elements)
    g1[6] = 0u;
    g1[7] = 0u;
    asm volatile("tensor_load_to_lds %0, %1" :: "s"(g0), "s"(g1) : "memory");
    __builtin_amdgcn_s_wait_tensorcnt(0);
  }
  __syncthreads();

  v8f acc = {0.f,0.f,0.f,0.f,0.f,0.f,0.f,0.f};
#pragma unroll
  for (int c0 = 0; c0 < CC; c0 += 32) {
    v16bf a  = load_A16(Wstk + (long)o0 * CC + c0, CC);
    v16bf bb = load_B16(smem + c0, CC);     // LDS reads
    acc = wmma_bf(a, bb, acc);
  }
  int lane = threadIdx.x & 31;
  int col = lane & 15, mo = (lane >> 4) ? 8 : 0;
#pragma unroll
  for (int r = 0; r < 8; ++r) {
    int o = o0 + r + mo;
    float v = alpha[o] * acc[r] + beta[o];
    v = v > 0.f ? v : 0.f;
    act[((long)b * OSTK + o) * NN + n0 + col] = f2bf(v);
  }
}

// ========== 7. theta softmax over N (in-place, block per (b,c)) ==========
__global__ void dgr_theta_sm(__bf16* __restrict__ act) {
  int c = blockIdx.x, b = blockIdx.y, t = threadIdx.x;
  __bf16* row = act + ((long)b * OSTK + KK + c) * NN;
  __shared__ float red[256];
  float m = -1e30f;
  for (int n = t; n < NN; n += 256) m = fmaxf(m, bf2f(row[n]));
  red[t] = m; __syncthreads();
  for (int w = 128; w; w >>= 1) { if (t < w) red[t] = fmaxf(red[t], red[t + w]); __syncthreads(); }
  float mx = red[0]; __syncthreads();
  float s = 0.f;
  for (int n = t; n < NN; n += 256) s += __expf(bf2f(row[n]) - mx);
  red[t] = s; __syncthreads();
  for (int w = 128; w; w >>= 1) { if (t < w) red[t] += red[t + w]; __syncthreads(); }
  float inv = 1.f / red[0]; __syncthreads();
  for (int n = t; n < NN; n += 256) row[n] = f2bf(__expf(bf2f(row[n]) - mx) * inv);
}

// ========== 8. rou softmax over C -> transposed rouT[b][n][c] ==========
__global__ void dgr_rou_sm(const __bf16* __restrict__ act, __bf16* __restrict__ rouT) {
  long i = blockIdx.x * (long)blockDim.x + threadIdx.x;
  if (i >= (long)BB * NN) return;
  int b = (int)(i / NN), n = (int)(i % NN);
  const __bf16* col = act + ((long)b * OSTK + KK + CC) * NN + n;
  float m = -1e30f;
  for (int c = 0; c < CC; ++c) m = fmaxf(m, bf2f(col[(long)c * NN]));
  float s = 0.f;
  for (int c = 0; c < CC; ++c) s += __expf(bf2f(col[(long)c * NN]) - m);
  float inv = 1.f / s;
  __bf16* out = rouT + ((long)b * NN + n) * CC;
  for (int c = 0; c < CC; ++c)
    out[c] = f2bf(__expf(bf2f(col[(long)c * NN]) - m) * inv);
}

// ========== 9. discrib split-K: partial[s][b][k][c] over n-chunk s ==========
__global__ void dgr_discrib(const __bf16* __restrict__ act, float* __restrict__ dscp) {
  int k0 = blockIdx.x * 16, c0 = blockIdx.y * 16;
  int b = blockIdx.z >> 3, s = blockIdx.z & 7;
  const __bf16* phi = act + (long)b * OSTK * NN;
  const __bf16* th  = act + ((long)b * OSTK + KK) * NN;
  v8f acc = {0.f,0.f,0.f,0.f,0.f,0.f,0.f,0.f};
  for (int n0 = s * NCHUNK; n0 < (s + 1) * NCHUNK; n0 += 32)
    acc = wmma_bf(load_A16(phi + (long)k0 * NN + n0, NN),
                  load_B16(th + (long)c0 * NN + n0, NN), acc);
  int lane = threadIdx.x & 31;
  int col = lane & 15, mo = (lane >> 4) ? 8 : 0;
#pragma unroll
  for (int r = 0; r < 8; ++r)
    dscp[(long)s * BB * KK * CC + ((long)b * KK + k0 + r + mo) * CC + c0 + col] = acc[r];
}
__global__ void dgr_dscred(const float* __restrict__ dscp, __bf16* __restrict__ dsc) {
  long i = blockIdx.x * (long)blockDim.x + threadIdx.x;
  if (i >= (long)BB * KK * CC) return;
  float s = 0.f;
  for (int p = 0; p < NSPLIT; ++p) s += dscp[(long)p * BB * KK * CC + i];
  dsc[i] = f2bf(s);
}

// ========== 10. Q[n][k] = sum_c discrib[k,c]*rou_sm[c,n] ==========
__global__ void dgr_qgemm(const __bf16* __restrict__ dsc, const __bf16* __restrict__ rouT,
                          float* __restrict__ Q) {
  int wave = threadIdx.x >> 5;
  int k0 = blockIdx.x * 16;
  int n0 = (blockIdx.y * 4 + wave) * 16;
  int b = blockIdx.z;
  const __bf16* A  = dsc + ((long)b * KK + k0) * CC;
  const __bf16* Bb = rouT + ((long)b * NN + n0) * CC;
  v8f acc = {0.f,0.f,0.f,0.f,0.f,0.f,0.f,0.f};
#pragma unroll
  for (int c0 = 0; c0 < CC; c0 += 32)
    acc = wmma_bf(load_A16(A + c0, CC), load_B16(Bb + c0, CC), acc);
  int lane = threadIdx.x & 31;
  int col = lane & 15, mo = (lane >> 4) ? 8 : 0;
#pragma unroll
  for (int r = 0; r < 8; ++r)
    Q[((long)b * NN + n0 + col) * KK + k0 + r + mo] = acc[r];
}

// ========== 11. L2 column norm of Q over N ==========
__global__ void dgr_qcolnorm(const float* __restrict__ Q, float* __restrict__ invn) {
  int k = blockIdx.x, b = blockIdx.y, t = threadIdx.x;
  float s = 0.f;
  for (int n = t; n < NN; n += 256) {
    float v = Q[((long)b * NN + n) * KK + k];
    s += v * v;
  }
  __shared__ float red[256];
  red[t] = s; __syncthreads();
  for (int w = 128; w; w >>= 1) { if (t < w) red[t] += red[t + w]; __syncthreads(); }
  if (t == 0) invn[b * KK + k] = 1.f / fmaxf(sqrtf(red[0]), 1e-12f);
}

// ========== 12. Q softmax over K; f32 in place + bf16 + bf16^T ==========
__global__ void dgr_qsoftmax(float* __restrict__ Q, const float* __restrict__ invn,
                             __bf16* __restrict__ Qbf, __bf16* __restrict__ QbfT) {
  long i = blockIdx.x * (long)blockDim.x + threadIdx.x;
  if (i >= (long)BB * NN) return;
  int b = (int)(i / NN), n = (int)(i % NN);
  float* row = Q + ((long)b * NN + n) * KK;
  const float* iv = invn + b * KK;
  float m = -1e30f;
  for (int k = 0; k < KK; ++k) m = fmaxf(m, row[k] * iv[k]);
  float s = 0.f;
  for (int k = 0; k < KK; ++k) s += __expf(row[k] * iv[k] - m);
  float is = 1.f / s;
  for (int k = 0; k < KK; ++k) {
    float v = __expf(row[k] * iv[k] - m) * is;
    row[k] = v;
    Qbf[((long)b * NN + n) * KK + k] = f2bf(v);
    QbfT[((long)b * KK + k) * NN + n] = f2bf(v);
  }
}

// ========== 13. Qsum[b][k] = sum_n Q[n][k] ==========
__global__ void dgr_qsum(const float* __restrict__ Q, float* __restrict__ Qsum) {
  int k = blockIdx.x, b = blockIdx.y, t = threadIdx.x;
  float s = 0.f;
  for (int n = t; n < NN; n += 256) s += Q[((long)b * NN + n) * KK + k];
  __shared__ float red[256];
  red[t] = s; __syncthreads();
  for (int w = 128; w; w >>= 1) { if (t < w) red[t] += red[t + w]; __syncthreads(); }
  if (t == 0) Qsum[b * KK + k] = red[0];
}

// ========== 14. Z split-K partials: Zp[s][b][c][k] ==========
__global__ void dgr_zgemm(const __bf16* __restrict__ act, const __bf16* __restrict__ QbfT,
                          float* __restrict__ Zp) {
  int c0 = blockIdx.x * 16, k0 = blockIdx.y * 16;
  int b = blockIdx.z >> 3, s = blockIdx.z & 7;
  const __bf16* A  = act + ((long)b * OSTK + KK + 2 * CC + c0) * NN;
  const __bf16* Bb = QbfT + ((long)b * KK + k0) * NN;
  v8f acc = {0.f,0.f,0.f,0.f,0.f,0.f,0.f,0.f};
  for (int n0 = s * NCHUNK; n0 < (s + 1) * NCHUNK; n0 += 32)
    acc = wmma_bf(load_A16(A + n0, NN), load_B16(Bb + n0, NN), acc);
  int lane = threadIdx.x & 31;
  int col = lane & 15, mo = (lane >> 4) ? 8 : 0;
#pragma unroll
  for (int r = 0; r < 8; ++r)
    Zp[(long)s * BB * CC * KK + ((long)b * CC + c0 + r + mo) * KK + k0 + col] = acc[r];
}
__global__ void dgr_zred(const float* __restrict__ Zp, const float* __restrict__ Qsum,
                         float* __restrict__ Z) {
  long i = blockIdx.x * (long)blockDim.x + threadIdx.x;
  if (i >= (long)BB * CC * KK) return;
  int k = (int)(i % KK);
  int b = (int)(i / ((long)CC * KK));
  float s = 0.f;
  for (int p = 0; p < NSPLIT; ++p) s += Zp[(long)p * BB * CC * KK + i];
  Z[i] = s / Qsum[b * KK + k];
}

// ========== 15. Z l2norm over C (in place) + g = Z^T as bf16 ==========
__global__ void dgr_znorm(float* __restrict__ Z, __bf16* __restrict__ gbf) {
  int k = blockIdx.x, b = blockIdx.y, t = threadIdx.x;  // t == c
  float v = Z[((long)b * CC + t) * KK + k];
  __shared__ float red[256];
  red[t] = v * v; __syncthreads();
  for (int w = 128; w; w >>= 1) { if (t < w) red[t] += red[t + w]; __syncthreads(); }
  float inv = 1.f / fmaxf(sqrtf(red[0]), 1e-12f);
  float z = v * inv;
  Z[((long)b * CC + t) * KK + k] = z;
  gbf[((long)b * KK + k) * CC + t] = f2bf(z);
}

// ========== 16. adj_raw[k][j] = sum_c g[k,c]*g[j,c] ==========
__global__ void dgr_adj(const __bf16* __restrict__ gbf, float* __restrict__ adjr) {
  int k0 = blockIdx.x * 16, j0 = blockIdx.y * 16, b = blockIdx.z;
  const __bf16* base = gbf + (long)b * KK * CC;
  v8f acc = {0.f,0.f,0.f,0.f,0.f,0.f,0.f,0.f};
#pragma unroll
  for (int c0 = 0; c0 < CC; c0 += 32)
    acc = wmma_bf(load_A16(base + (long)k0 * CC + c0, CC),
                  load_B16(base + (long)j0 * CC + c0, CC), acc);
  int lane = threadIdx.x & 31;
  int col = lane & 15, mo = (lane >> 4) ? 8 : 0;
#pragma unroll
  for (int r = 0; r < 8; ++r)
    adjr[((long)b * KK + k0 + r + mo) * KK + j0 + col] = acc[r];
}

// ========== 17. adj row softmax -> bf16 ==========
__global__ void dgr_adjsm(const float* __restrict__ adjr, __bf16* __restrict__ adjbf) {
  int i = blockIdx.x * blockDim.x + threadIdx.x;
  if (i >= BB * KK) return;
  const float* row = adjr + (long)i * KK;
  __bf16* orow = adjbf + (long)i * KK;
  float m = -1e30f;
  for (int j = 0; j < KK; ++j) m = fmaxf(m, row[j]);
  float s = 0.f;
  for (int j = 0; j < KK; ++j) s += __expf(row[j] - m);
  float is = 1.f / s;
  for (int j = 0; j < KK; ++j) orow[j] = f2bf(__expf(row[j] - m) * is);
}

// ========== 18. support^T[o][k] = sum_c g[k,c]*gcn_w[c,o] ==========
__global__ void dgr_support(const __bf16* __restrict__ gbf, const __bf16* __restrict__ gwT,
                            __bf16* __restrict__ supT) {
  int k0 = blockIdx.x * 16, o0 = blockIdx.y * 16, b = blockIdx.z;
  v8f acc = {0.f,0.f,0.f,0.f,0.f,0.f,0.f,0.f};
#pragma unroll
  for (int c0 = 0; c0 < CC; c0 += 32)
    acc = wmma_bf(load_A16(gbf + ((long)b * KK + k0) * CC + c0, CC),
                  load_B16(gwT + (long)o0 * CC + c0, CC), acc);
  int lane = threadIdx.x & 31;
  int col = lane & 15, mo = (lane >> 4) ? 8 : 0;
#pragma unroll
  for (int r = 0; r < 8; ++r)
    supT[((long)b * CC + o0 + col) * KK + k0 + r + mo] = f2bf(acc[r]);
}

// ========== 19. gout_pre[c][k] = relu(sum_j adj[k,j]*support[j,c]) ==========
__global__ void dgr_sgemm(const __bf16* __restrict__ adjbf, const __bf16* __restrict__ supT,
                          float* __restrict__ goutp) {
  int k0 = blockIdx.x * 16, c0 = blockIdx.y * 16, b = blockIdx.z;
  v8f acc = {0.f,0.f,0.f,0.f,0.f,0.f,0.f,0.f};
#pragma unroll
  for (int j0 = 0; j0 < KK; j0 += 32)
    acc = wmma_bf(load_A16(adjbf + ((long)b * KK + k0) * KK + j0, KK),
                  load_B16(supT + ((long)b * CC + c0) * KK + j0, KK), acc);
  int lane = threadIdx.x & 31;
  int col = lane & 15, mo = (lane >> 4) ? 8 : 0;
#pragma unroll
  for (int r = 0; r < 8; ++r) {
    int c = c0 + col, k = k0 + r + mo;
    goutp[((long)b * CC + c) * KK + k] = fmaxf(acc[r], 0.f);
  }
}

// ========== 20. BN stats of gout over (b,k) ==========
__global__ void dgr_goutstats(const float* __restrict__ goutp, const float* gs,
                              const float* gb, float* ag, float* bg) {
  int c = blockIdx.x, t = threadIdx.x;
  float s = 0.f, s2 = 0.f;
  for (int i = t; i < BB * KK; i += 256) {
    int b = i >> 6, k = i & 63;
    float v = goutp[((long)b * CC + c) * KK + k];
    s += v; s2 += v * v;
  }
  __shared__ float r1[256], r2[256];
  r1[t] = s; r2[t] = s2; __syncthreads();
  for (int w = 128; w; w >>= 1) { if (t < w) { r1[t] += r1[t + w]; r2[t] += r2[t + w]; } __syncthreads(); }
  if (t == 0) {
    float cnt = (float)(BB * KK);
    float mu = r1[0] / cnt;
    float var = r2[0] / cnt - mu * mu;
    if (var < 0.f) var = 0.f;
    float a = gs[c] * rsqrtf(var + 1e-5f);
    ag[c] = a; bg[c] = gb[c] - mu * a;
  }
}

// ========== 21. graph = BN(gout)*gamma + Zn (f32 + bf16) ==========
__global__ void dgr_graph(const float* __restrict__ goutp, const float* __restrict__ ag,
                          const float* __restrict__ bg, const float* __restrict__ gamma,
                          const float* __restrict__ Z, float* __restrict__ graph,
                          __bf16* __restrict__ graphbf) {
  long i = blockIdx.x * (long)blockDim.x + threadIdx.x;
  if (i >= (long)BB * CC * KK) return;
  int c = (int)((i / KK) % CC);
  float g = ag[c] * goutp[i] + bg[c];
  float v = g * gamma[0] + Z[i];
  graph[i] = v; graphbf[i] = f2bf(v);
}

// ========== 22. QTQ[b][k1][k2] = sum_n Q[n,k1]*Q[n,k2] ==========
__global__ void dgr_qtq(const float* __restrict__ Q, float* __restrict__ QTQ) {
  int k1 = blockIdx.x, b = blockIdx.y, k2 = threadIdx.x;
  float s = 0.f;
  const float* base = Q + (long)b * NN * KK;
  for (int n = 0; n < NN; ++n) {
    const float* row = base + (long)n * KK;
    s += row[k1] * row[k2];
  }
  QTQ[((long)b * KK + k1) * KK + k2] = s;
}

// ========== 23. analytic BN of final output ==========
__global__ void dgr_outstats(const float* __restrict__ graph, const float* __restrict__ Qsum,
                             const float* __restrict__ QTQ, const float* os, const float* ob,
                             float* ao, float* bo) {
  int c = blockIdx.x, k1 = threadIdx.x;  // 64 threads
  float mu_p = 0.f, m2_p = 0.f;
  for (int b = 0; b < BB; ++b) {
    float g1 = graph[((long)b * CC + c) * KK + k1];
    mu_p += g1 * Qsum[b * KK + k1];
    float inner = 0.f;
    for (int k2 = 0; k2 < KK; ++k2)
      inner += QTQ[((long)b * KK + k1) * KK + k2] * graph[((long)b * CC + c) * KK + k2];
    m2_p += g1 * inner;
  }
  __shared__ float r1[64], r2[64];
  r1[k1] = mu_p; r2[k1] = m2_p; __syncthreads();
  for (int w = 32; w; w >>= 1) { if (k1 < w) { r1[k1] += r1[k1 + w]; r2[k1] += r2[k1 + w]; } __syncthreads(); }
  if (k1 == 0) {
    float mu = r1[0] / BN_CNT;
    float var = r2[0] / BN_CNT - mu * mu;
    if (var < 0.f) var = 0.f;
    float a = os[c] * rsqrtf(var + 1e-4f);
    ao[c] = a; bo[c] = ob[c] - mu * a;
  }
}

// ========== 24. fused out GEMM + BN + residual -> d_out ==========
__global__ void dgr_outgemm(const __bf16* __restrict__ graphbf, const __bf16* __restrict__ Qbf,
                            const float* __restrict__ ao, const float* __restrict__ bo,
                            const float* __restrict__ x, float* __restrict__ out) {
  int wave = threadIdx.x >> 5;
  int c0 = blockIdx.x * 16;
  int n0 = (blockIdx.y * 4 + wave) * 16;
  int b = blockIdx.z;
  const __bf16* A  = graphbf + ((long)b * CC + c0) * KK;
  const __bf16* Bb = Qbf + ((long)b * NN + n0) * KK;
  __builtin_prefetch(Bb, 0, 1);
  v8f acc = {0.f,0.f,0.f,0.f,0.f,0.f,0.f,0.f};
#pragma unroll
  for (int k0 = 0; k0 < KK; k0 += 32)
    acc = wmma_bf(load_A16(A + k0, KK), load_B16(Bb + k0, KK), acc);
  int lane = threadIdx.x & 31;
  int col = lane & 15, mo = (lane >> 4) ? 8 : 0;
#pragma unroll
  for (int r = 0; r < 8; ++r) {
    int c = c0 + r + mo;
    long idx = ((long)b * CC + c) * NN + n0 + col;
    out[idx] = ao[c] * acc[r] + bo[c] + x[idx];
  }
}

extern "C" void kernel_launch(void* const* d_in, const int* in_sizes, int n_in,
                              void* d_out, int out_size, void* d_ws, size_t ws_size,
                              hipStream_t stream) {
  (void)in_sizes; (void)n_in; (void)out_size; (void)ws_size;
  const float* x    = (const float*)d_in[0];
  const float* wphi = (const float*)d_in[1];
  const float* sphi = (const float*)d_in[2];
  const float* bphi = (const float*)d_in[3];
  const float* wth  = (const float*)d_in[4];
  const float* sth  = (const float*)d_in[5];
  const float* bth  = (const float*)d_in[6];
  const float* wro  = (const float*)d_in[7];
  const float* sro  = (const float*)d_in[8];
  const float* bro  = (const float*)d_in[9];
  const float* wva  = (const float*)d_in[10];
  const float* sva  = (const float*)d_in[11];
  const float* bva  = (const float*)d_in[12];
  const float* gcnw = (const float*)d_in[13];
  const float* gcns = (const float*)d_in[14];
  const float* gcnb = (const float*)d_in[15];
  const float* gamma= (const float*)d_in[16];
  const float* outs = (const float*)d_in[17];
  const float* outb = (const float*)d_in[18];
  float* out = (float*)d_out;

  char* wsp = (char*)d_ws; size_t off = 0;
  auto carve = [&](size_t bytes) -> char* {
    char* p = wsp + off;
    off = (off + bytes + 255) & ~(size_t)255;
    return p;
  };
  __bf16* xbT     = (__bf16*)carve((size_t)BB * NN * CC * 2);
  __bf16* Wstk    = (__bf16*)carve((size_t)OSTK * CC * 2);
  __bf16* gwT     = (__bf16*)carve((size_t)CC * CC * 2);
  float*  Cxxb    = (float*) carve((size_t)BB * CC * CC * 4);
  float*  Cxx     = (float*) carve((size_t)CC * CC * 4);
  float*  xmean   = (float*) carve((size_t)CC * 4);
  float*  alpha   = (float*) carve((size_t)OSTK * 4);
  float*  beta    = (float*) carve((size_t)OSTK * 4);
  __bf16* act     = (__bf16*)carve((size_t)BB * OSTK * NN * 2);
  __bf16* rouT    = (__bf16*)carve((size_t)BB * NN * CC * 2);
  float*  dscp    = (float*) carve((size_t)NSPLIT * BB * KK * CC * 4);
  __bf16* dsc     = (__bf16*)carve((size_t)BB * KK * CC * 2);
  float*  Q       = (float*) carve((size_t)BB * NN * KK * 4);
  __bf16* Qbf     = (__bf16*)carve((size_t)BB * NN * KK * 2);
  __bf16* QbfT    = (__bf16*)carve((size_t)BB * KK * NN * 2);
  float*  invn    = (float*) carve((size_t)BB * KK * 4);
  float*  Qsum    = (float*) carve((size_t)BB * KK * 4);
  float*  Zp      = (float*) carve((size_t)NSPLIT * BB * CC * KK * 4);
  float*  Z       = (float*) carve((size_t)BB * CC * KK * 4);
  __bf16* gbf     = (__bf16*)carve((size_t)BB * KK * CC * 2);
  float*  adjr    = (float*) carve((size_t)BB * KK * KK * 4);
  __bf16* adjbf   = (__bf16*)carve((size_t)BB * KK * KK * 2);
  __bf16* supT    = (__bf16*)carve((size_t)BB * CC * KK * 2);
  float*  goutp   = (float*) carve((size_t)BB * CC * KK * 4);
  float*  ag      = (float*) carve((size_t)CC * 4);
  float*  bg      = (float*) carve((size_t)CC * 4);
  float*  graph   = (float*) carve((size_t)BB * CC * KK * 4);
  __bf16* graphbf = (__bf16*)carve((size_t)BB * CC * KK * 2);
  float*  QTQ     = (float*) carve((size_t)BB * KK * KK * 4);
  float*  ao      = (float*) carve((size_t)CC * 4);
  float*  bo      = (float*) carve((size_t)CC * 4);

  dgr_xT<<<1024, 256, 0, stream>>>(x, xbT);
  dgr_wpack<<<512, 256, 0, stream>>>(wphi, wth, wro, wva, gcnw, Wstk, gwT);
  dgr_xmean<<<CC, 256, 0, stream>>>(x, xmean);
  dgr_cxx<<<dim3(16, 16, BB), 32, 0, stream>>>(x, Cxxb);
  dgr_cxxred<<<256, 256, 0, stream>>>(Cxxb, Cxx);
  dgr_bnparams<<<OSTK, 256, 0, stream>>>(wphi, wth, wro, wva, sphi, bphi, sth, bth,
                                         sro, bro, sva, bva, Cxx, xmean, alpha, beta);
  dgr_proj<<<dim3(13, NN / 16, BB), 128, 0, stream>>>(Wstk, xbT, alpha, beta, act);
  dgr_theta_sm<<<dim3(CC, BB), 256, 0, stream>>>(act);
  dgr_rou_sm<<<(BB * NN + 255) / 256, 256, 0, stream>>>(act, rouT);
  dgr_discrib<<<dim3(KK / 16, CC / 16, BB * NSPLIT), 32, 0, stream>>>(act, dscp);
  dgr_dscred<<<(BB * KK * CC + 255) / 256, 256, 0, stream>>>(dscp, dsc);
  dgr_qgemm<<<dim3(KK / 16, NN / 64, BB), 128, 0, stream>>>(dsc, rouT, Q);
  dgr_qcolnorm<<<dim3(KK, BB), 256, 0, stream>>>(Q, invn);
  dgr_qsoftmax<<<(BB * NN + 255) / 256, 256, 0, stream>>>(Q, invn, Qbf, QbfT);
  dgr_qsum<<<dim3(KK, BB), 256, 0, stream>>>(Q, Qsum);
  dgr_zgemm<<<dim3(CC / 16, KK / 16, BB * NSPLIT), 32, 0, stream>>>(act, QbfT, Zp);
  dgr_zred<<<(BB * CC * KK + 255) / 256, 256, 0, stream>>>(Zp, Qsum, Z);
  dgr_znorm<<<dim3(KK, BB), 256, 0, stream>>>(Z, gbf);
  dgr_adj<<<dim3(KK / 16, KK / 16, BB), 32, 0, stream>>>(gbf, adjr);
  dgr_adjsm<<<(BB * KK + 255) / 256, 256, 0, stream>>>(adjr, adjbf);
  dgr_support<<<dim3(KK / 16, CC / 16, BB), 32, 0, stream>>>(gbf, gwT, supT);
  dgr_sgemm<<<dim3(KK / 16, CC / 16, BB), 32, 0, stream>>>(adjbf, supT, goutp);
  dgr_goutstats<<<CC, 256, 0, stream>>>(goutp, gcns, gcnb, ag, bg);
  dgr_graph<<<(BB * CC * KK + 255) / 256, 256, 0, stream>>>(goutp, ag, bg, gamma, Z,
                                                            graph, graphbf);
  dgr_qtq<<<dim3(KK, BB), KK, 0, stream>>>(Q, QTQ);
  dgr_outstats<<<CC, KK, 0, stream>>>(graph, Qsum, QTQ, outs, outb, ao, bo);
  dgr_outgemm<<<dim3(CC / 16, NN / 64, BB), 128, 0, stream>>>(graphbf, Qbf, ao, bo, x, out);
}
